// GNN_26843545600638
// MI455X (gfx1250) — compile-verified, matrix operand order checked by reading, not run on hardware
//
#include <hip/hip_runtime.h>
#include <hip/hip_bf16.h>
#include <math.h>

// ---------------------------------------------------------------------------
// CDNA5 (gfx1250) GCN forward. wave32. WMMA bf16 16x16x32, f32 accumulate.
// One wave computes a 16-row x (NTILES*16)-col output strip: A fragment loaded
// once per k-step and reused across NTILES back-to-back v_wmma ops.
// ---------------------------------------------------------------------------

typedef __attribute__((ext_vector_type(16))) __bf16 v16bf;
typedef __attribute__((ext_vector_type(8)))  float  v8f;

#define DH 64

// ---------------- utility kernels ----------------

__global__ void zero_f32(float* __restrict__ p, int n) {
    int i = blockIdx.x * blockDim.x + threadIdx.x;
    if (i < n) p[i] = 0.0f;
}

__global__ void deg_count(const int* __restrict__ dst, int E, float* __restrict__ deg) {
    int e = blockIdx.x * blockDim.x + threadIdx.x;
    if (e < E) atomicAdd(&deg[dst[e]], 1.0f);
}

// in place: deg -> dinv = rsqrt(deg + 1)   (+1 = self loop)
__global__ void make_dinv(float* __restrict__ d, int n) {
    int i = blockIdx.x * blockDim.x + threadIdx.x;
    if (i < n) d[i] = rsqrtf(d[i] + 1.0f);
}

// ---------------- WMMA GEMM: OUT[M x NC] = X[M x K] @ W[K x NC] --------------
// K, NTILES compile-time; NC = NTILES*16. One wave -> rows [mt*16, mt*16+16),
// all NC columns. A-fragment lane layout (16-bit A 16x32, ISA 7.12.2): lane l:
// row = l&15, half h = l>>4; VGPR v<4 holds K pair (8h+2v, 8h+2v+1), v>=4
// holds (16+8h+2(v-4), +1). B mirrors A with column = l&15. C/D: row = v+8h,
// col = l&15.
template <int K, int NTILES>
__global__ void gemm_wmma_bf16(const float* __restrict__ X,
                               const float* __restrict__ W,
                               float* __restrict__ OUT,
                               int mtiles, int Nrows) {
    constexpr int NC = NTILES * 16;
    int gid  = blockIdx.x * blockDim.x + threadIdx.x;
    int wave = gid >> 5;
    int lane = threadIdx.x & 31;
    if (wave >= mtiles) return;                 // uniform per-wave: EXEC stays all-1
    int mt = wave;

    int m  = lane & 15;
    int hf = lane >> 4;
    int arow = mt * 16 + m;
    if (arow >= Nrows) arow = Nrows - 1;        // clamp loads (stores handled below)

    v8f acc[NTILES];
#pragma unroll
    for (int t = 0; t < NTILES; ++t) acc[t] = (v8f){};

#pragma unroll
    for (int k0 = 0; k0 < K; k0 += 32) {
        v16bf a;
#pragma unroll
        for (int v = 0; v < 8; ++v) {
            int kk = k0 + ((v < 4) ? (8 * hf + 2 * v) : (16 + 8 * hf + 2 * (v - 4)));
            float a0 = X[arow * K + kk];
            float a1 = X[arow * K + kk + 1];
            a[2 * v]     = (__bf16)a0;
            a[2 * v + 1] = (__bf16)a1;
        }
#pragma unroll
        for (int t = 0; t < NTILES; ++t) {
            int col = t * 16 + m;
            v16bf b;
#pragma unroll
            for (int v = 0; v < 8; ++v) {
                int kk = k0 + ((v < 4) ? (8 * hf + 2 * v) : (16 + 8 * hf + 2 * (v - 4)));
                float b0 = W[kk * NC + col];
                float b1 = W[(kk + 1) * NC + col];
                b[2 * v]     = (__bf16)b0;
                b[2 * v + 1] = (__bf16)b1;
            }
            acc[t] = __builtin_amdgcn_wmma_f32_16x16x32_bf16(
                /*neg_a=*/false, a, /*neg_b=*/false, b,
                /*c_mod=*/(short)0, acc[t], /*reuse_a=*/false, /*reuse_b=*/false);
        }
    }

    int row0 = mt * 16;
    if (row0 + 16 <= Nrows) {                   // uniform fast path (always, N%16==0)
#pragma unroll
        for (int t = 0; t < NTILES; ++t) {
            float* o = OUT + row0 * NC + t * 16 + m;
#pragma unroll
            for (int v = 0; v < 8; ++v)
                o[(v + 8 * hf) * NC] = acc[t][v];
        }
    } else {                                    // generic tail (not hit for N=50000)
#pragma unroll
        for (int t = 0; t < NTILES; ++t)
#pragma unroll
            for (int v = 0; v < 8; ++v) {
                int r = row0 + v + 8 * hf;
                if (r < Nrows) OUT[r * NC + t * 16 + m] = acc[t][v];
            }
    }
}

// ---------------- edge scatter: acc[dst] += t[src] * dinv[src]*dinv[dst] ------
// 4 threads per edge, 16 features each, float4 gathers, f32 atomics (L2-resident
// accumulator: 12.8 MB << 192 MB L2).
__global__ void scatter_edges(const int* __restrict__ src, const int* __restrict__ dst,
                              int E, const float* __restrict__ dinv,
                              const float* __restrict__ t, float* __restrict__ acc) {
    int gid = blockIdx.x * blockDim.x + threadIdx.x;
    int e = gid >> 2;
    int c = gid & 3;
    if (e >= E) return;
    int s = src[e];
    int d = dst[e];
    float nrm = dinv[s] * dinv[d];
    const float4* tp = (const float4*)(t + s * DH + c * 16);
    float* ap = acc + d * DH + c * 16;
#pragma unroll
    for (int q = 0; q < 4; ++q) {
        float4 v = tp[q];
        atomicAdd(ap + 4 * q + 0, v.x * nrm);
        atomicAdd(ap + 4 * q + 1, v.y * nrm);
        atomicAdd(ap + 4 * q + 2, v.z * nrm);
        atomicAdd(ap + 4 * q + 3, v.w * nrm);
    }
}

// ---------------- epilogue: self-loop + bias + ELU (in place on acc) ----------
__global__ void bias_selfloop_elu(float* __restrict__ acc, const float* __restrict__ t,
                                  const float* __restrict__ dinv,
                                  const float* __restrict__ bias, int Nn) {
    int idx = blockIdx.x * blockDim.x + threadIdx.x;
    if (idx >= Nn * DH) return;
    int row = idx >> 6;          // DH = 64
    int j   = idx & (DH - 1);
    float di = dinv[row];
    float h = acc[idx] + t[idx] * di * di + bias[j];
    acc[idx] = (h > 0.0f) ? h : (__expf(h) - 1.0f);
}

// ---------------- softmax over DOUT=16, one thread per row --------------------
__global__ void fc_bias_softmax(const float* __restrict__ logits,
                                const float* __restrict__ fcb,
                                float* __restrict__ out, int Nn) {
    int row = blockIdx.x * blockDim.x + threadIdx.x;
    if (row >= Nn) return;
    float v[16];
    float mx = -3.0e38f;
#pragma unroll
    for (int j = 0; j < 16; ++j) {
        v[j] = logits[row * 16 + j] + fcb[j];
        mx = fmaxf(mx, v[j]);
    }
    float sum = 0.0f;
#pragma unroll
    for (int j = 0; j < 16; ++j) {
        v[j] = __expf(v[j] - mx);
        sum += v[j];
    }
    float inv = 1.0f / sum;
#pragma unroll
    for (int j = 0; j < 16; ++j)
        out[row * 16 + j] = v[j] * inv;
}

// ---------------------------------------------------------------------------
extern "C" void kernel_launch(void* const* d_in, const int* in_sizes, int n_in,
                              void* d_out, int out_size, void* d_ws, size_t ws_size,
                              hipStream_t stream) {
    const float* x   = (const float*)d_in[0];
    const int*  eidx = (const int*)  d_in[1];
    const float* w0  = (const float*)d_in[2];
    const float* b0  = (const float*)d_in[3];
    const float* w1  = (const float*)d_in[4];
    const float* b1  = (const float*)d_in[5];
    const float* w2  = (const float*)d_in[6];
    const float* b2  = (const float*)d_in[7];
    const float* fcw = (const float*)d_in[8];
    const float* fcb = (const float*)d_in[9];

    const int DIN = 128;
    const int N = in_sizes[0] / DIN;    // 50000
    const int E = in_sizes[1] / 2;      // 800000
    const int* srcI = eidx;
    const int* dstI = eidx + E;

    // workspace layout (floats): dinv[N] | bufA[N*64] | bufB[N*64] | bufC[N*64]
    float* dinv = (float*)d_ws;
    float* bufA = dinv + N;
    float* bufB = bufA + (size_t)N * DH;
    float* bufC = bufB + (size_t)N * DH;

    const int B = 256;
    const int NDH = N * DH;
    const int mtiles = (N + 15) / 16;
    const int gthr = mtiles * 32;                 // one wave per 16-row strip

    // ---- degrees -> dinv ----
    zero_f32 <<<(N + B - 1) / B, B, 0, stream>>>(dinv, N);
    deg_count<<<(E + B - 1) / B, B, 0, stream>>>(dstI, E, dinv);
    make_dinv<<<(N + B - 1) / B, B, 0, stream>>>(dinv, N);

    auto aggregate = [&](float* t, float* accb, const float* bias) {
        zero_f32<<<(NDH + B - 1) / B, B, 0, stream>>>(accb, NDH);
        scatter_edges<<<(E * 4 + B - 1) / B, B, 0, stream>>>(srcI, dstI, E, dinv, t, accb);
        bias_selfloop_elu<<<(NDH + B - 1) / B, B, 0, stream>>>(accb, t, dinv, bias, N);
    };

    // layer 1: x (K=128) -> bufB
    gemm_wmma_bf16<128, 4><<<(gthr + B - 1) / B, B, 0, stream>>>(x, w0, bufA, mtiles, N);
    aggregate(bufA, bufB, b0);
    // layer 2: bufB (K=64) -> bufC
    gemm_wmma_bf16<64, 4><<<(gthr + B - 1) / B, B, 0, stream>>>(bufB, w1, bufA, mtiles, N);
    aggregate(bufA, bufC, b1);
    // layer 3: bufC (K=64) -> bufB
    gemm_wmma_bf16<64, 4><<<(gthr + B - 1) / B, B, 0, stream>>>(bufC, w2, bufA, mtiles, N);
    aggregate(bufA, bufB, b2);

    // FC (64 -> 16) + softmax
    gemm_wmma_bf16<64, 1><<<(gthr + B - 1) / B, B, 0, stream>>>(bufB, fcw, bufC, mtiles, N);
    fc_bias_softmax<<<(N + B - 1) / B, B, 0, stream>>>(bufC, fcb, (float*)d_out, N);
}
